// CRFLayer_28183575396909
// MI455X (gfx1250) — matrix operand order repaired and checked by reading, not compiled, not run on hardware
//
#include <hip/hip_runtime.h>
#include <hip/hip_bf16.h>

typedef __attribute__((ext_vector_type(16))) _Float16 v16h;
typedef __attribute__((ext_vector_type(8)))  float    v8f;

#define CRF_B 4096
#define CRF_S 512
#define CRF_T 32
#define CRF_PF 16   // prefetch distance in sequence steps (16*128B = 2KB/batch)

// ---------------------------------------------------------------------------
// Forward algorithm in scaled-exp space.
// Wave32 handles 16 batches. State: Bp (B-operand layout, 32x16 f16):
//   lane m      holds p[m][k], k=0..15  (16 packed halves)
//   lane m+16   holds p[m][k], k=16..31
// Static A tiles: A0[n][k] = exp(trans[k][n]-tmax) for n=0..15, A1 for n=16..31
// Per step:  D = A x Bp  ->  lane m: q[m][n] n=0..7 (D0), 16..23 (D1)
//                            lane m+16:      n=8..15 (D0), 24..31 (D1)
// u = D * exp(em);  w = max_n u;  p' = u/w;  c += tmax + log(w)
// ---------------------------------------------------------------------------
__global__ __launch_bounds__(32) void crf_forward_kernel(
    const float* __restrict__ emissions,
    const unsigned char* __restrict__ mask,
    const float* __restrict__ transitions,
    const float* __restrict__ start_t,
    const float* __restrict__ end_t,
    float* __restrict__ fwd_out)
{
    const int  lane = threadIdx.x & 31;
    const int  m    = lane & 15;
    const bool hi   = lane >= 16;
    const int  b    = blockIdx.x * 16 + m;

    // ---- global shift for transitions ----
    float tmax = -1e30f;
    #pragma unroll
    for (int k = 0; k < CRF_T; ++k)
        tmax = fmaxf(tmax, transitions[k * CRF_T + lane]);
    #pragma unroll
    for (int o = 16; o >= 1; o >>= 1)
        tmax = fmaxf(tmax, __shfl_xor(tmax, o, 32));

    // ---- build static A tiles (expT transposed, f16, WMMA A layout) ----
    // lane L: M = L%16; half i -> K = (hi?16:0)+i
    v16h A0, A1;
    #pragma unroll
    for (int i = 0; i < 16; ++i) {
        int kk = (hi ? 16 : 0) + i;
        A0[i] = (_Float16)__expf(transitions[kk * CRF_T + m]        - tmax);
        A1[i] = (_Float16)__expf(transitions[kk * CRF_T + 16 + m]   - tmax);
    }

    // ---- initial distribution: score0 = start + em[:,0,:] ----
    const int nb = hi ? 16 : 0;
    float e0[16];
    {
        const float* em0 = emissions + (size_t)b * CRF_S * CRF_T + nb;
        #pragma unroll
        for (int i = 0; i < 16; ++i) e0[i] = start_t[nb + i] + em0[i];
    }
    float mx = -1e30f;
    #pragma unroll
    for (int i = 0; i < 16; ++i) mx = fmaxf(mx, e0[i]);
    mx = fmaxf(mx, __shfl_xor(mx, 16, 32));
    float c = mx;                       // running log-scale (same in lane pair)
    v16h Bp;
    #pragma unroll
    for (int i = 0; i < 16; ++i) Bp[i] = (_Float16)__expf(e0[i] - mx);

    union H8 { _Float16 h[8]; int i[4]; };

    const float*         emBase = emissions + (size_t)b * CRF_S * CRF_T + (hi ? 8 : 0);
    const unsigned char* mkBase = mask + (size_t)b * CRF_S;

    // ---- warm the stream: prefetch the first CRF_PF steps ----
    #pragma unroll
    for (int s = 1; s < 1 + CRF_PF; ++s)
        __builtin_prefetch(emBase + (size_t)s * CRF_T, 0, 1);
    __builtin_prefetch(mkBase, 0, 1);

    // ---- sequential chain over s = 1..S-1 ----
    for (int s = 1; s < CRF_S; ++s) {
        const float* emp = emBase + (size_t)s * CRF_T;
        float4 la0 = *(const float4*)(emp);
        float4 la1 = *(const float4*)(emp + 4);
        float4 lb0 = *(const float4*)(emp + 16);
        float4 lb1 = *(const float4*)(emp + 20);
        const unsigned char mk = mkBase[s];

        // stage future lines in cache (clamped in-bounds; no counter traffic)
        int spf = s + CRF_PF; spf = spf < CRF_S ? spf : (CRF_S - 1);
        __builtin_prefetch(emBase + (size_t)spf * CRF_T, 0, 1);

        v8f zero = {};
        v8f d0 = __builtin_amdgcn_wmma_f32_16x16x32_f16(false, A0, false, Bp,
                                                        (short)0, zero, false, false);
        v8f d1 = __builtin_amdgcn_wmma_f32_16x16x32_f16(false, A1, false, Bp,
                                                        (short)0, zero, false, false);

        float ea[8] = {la0.x, la0.y, la0.z, la0.w, la1.x, la1.y, la1.z, la1.w};
        float eb[8] = {lb0.x, lb0.y, lb0.z, lb0.w, lb1.x, lb1.y, lb1.z, lb1.w};

        float u0[8], u1[8];
        #pragma unroll
        for (int i = 0; i < 8; ++i) {
            u0[i] = d0[i] * __expf(ea[i]);
            u1[i] = d1[i] * __expf(eb[i]);
        }

        // per-batch max (lane pair m <-> m+16 shares one batch)
        float w = -1e30f;
        #pragma unroll
        for (int i = 0; i < 8; ++i) w = fmaxf(w, fmaxf(u0[i], u1[i]));
        w = fmaxf(w, __shfl_xor(w, 16, 32));
        float inv = 1.0f / w;

        H8 own0, own1, send, recv;
        #pragma unroll
        for (int i = 0; i < 8; ++i) {
            own0.h[i] = (_Float16)(u0[i] * inv);
            own1.h[i] = (_Float16)(u1[i] * inv);
        }
        send = hi ? own0 : own1;                 // lo sends n=16..23, hi sends n=8..15
        #pragma unroll
        for (int j = 0; j < 4; ++j) recv.i[j] = __shfl_xor(send.i[j], 16, 32);

        v16h np;
        #pragma unroll
        for (int i = 0; i < 8; ++i) {
            np[i]     = hi ? recv.h[i] : own0.h[i];   // k = 0..7   / 16..23
            np[i + 8] = hi ? own1.h[i] : recv.h[i];   // k = 8..15  / 24..31
        }

        // mask: keep old state when step is masked out (no branch: EXEC stays full)
        Bp = mk ? np : Bp;
        c += mk ? (tmax + __logf(w)) : 0.0f;
    }

    // ---- terminate: fwd = c + log( sum_k p[k] * exp(end[k]) ) ----
    float tsum = 0.0f;
    #pragma unroll
    for (int i = 0; i < 16; ++i)
        tsum += (float)Bp[i] * __expf(end_t[nb + i]);
    tsum += __shfl_xor(tsum, 16, 32);
    if (!hi) fwd_out[b] = c + __logf(tsum);
}

// ---------------------------------------------------------------------------
// Gold path score: one wave per batch, lanes stride the sequence.
// ---------------------------------------------------------------------------
__global__ __launch_bounds__(256) void crf_gold_kernel(
    const float* __restrict__ emissions,
    const int* __restrict__ tags,
    const unsigned char* __restrict__ mask,
    const float* __restrict__ transitions,
    const float* __restrict__ start_t,
    const float* __restrict__ end_t,
    float* __restrict__ gold_out)
{
    const int lane = threadIdx.x & 31;
    const int wave = threadIdx.x >> 5;
    const int b    = blockIdx.x * 8 + wave;

    const int*           tg = tags + (size_t)b * CRF_S;
    const unsigned char* mk = mask + (size_t)b * CRF_S;
    const float*         em = emissions + (size_t)b * CRF_S * CRF_T;

    float acc = 0.0f;
    int   cnt = 0;
    for (int s = 1 + lane; s < CRF_S; s += 32) {
        int   tp = tg[s - 1], tc = tg[s];
        float f  = mk[s] ? 1.0f : 0.0f;
        acc += f * (transitions[tp * CRF_T + tc] + em[(size_t)s * CRF_T + tc]);
        cnt += mk[s] ? 1 : 0;
    }
    #pragma unroll
    for (int o = 16; o >= 1; o >>= 1) {
        acc += __shfl_xor(acc, o, 32);
        cnt += __shfl_xor(cnt, o, 32);
    }
    if (lane == 0) {
        int t0       = tg[0];
        int last_idx = (cnt + (mk[0] ? 1 : 0)) - 1;   // mask.sum() - 1
        int lastTag  = tg[last_idx];
        gold_out[b] = start_t[t0] + em[t0] + acc + end_t[lastTag];
    }
}

// ---------------------------------------------------------------------------
// Deterministic mean of (fwd - gold): single block tree reduction.
// ---------------------------------------------------------------------------
__global__ __launch_bounds__(1024) void crf_reduce_kernel(
    const float* __restrict__ fwd,
    const float* __restrict__ gold,
    float* __restrict__ out)
{
    __shared__ float sm[1024];
    int t = threadIdx.x;
    float a = 0.0f;
    for (int i = t; i < CRF_B; i += 1024) a += fwd[i] - gold[i];
    sm[t] = a;
    __syncthreads();
    for (int o = 512; o >= 1; o >>= 1) {
        if (t < o) sm[t] += sm[t + o];
        __syncthreads();
    }
    if (t == 0) out[0] = sm[0] / (float)CRF_B;
}

extern "C" void kernel_launch(void* const* d_in, const int* in_sizes, int n_in,
                              void* d_out, int out_size, void* d_ws, size_t ws_size,
                              hipStream_t stream) {
    const float*         emissions   = (const float*)d_in[0];
    const int*           tags        = (const int*)d_in[1];
    const unsigned char* mask        = (const unsigned char*)d_in[2];
    const float*         transitions = (const float*)d_in[3];
    const float*         start_t     = (const float*)d_in[4];
    const float*         end_t       = (const float*)d_in[5];

    float* fwd  = (float*)d_ws;           // CRF_B floats
    float* gold = fwd + CRF_B;            // CRF_B floats

    crf_forward_kernel<<<CRF_B / 16, 32, 0, stream>>>(
        emissions, mask, transitions, start_t, end_t, fwd);
    crf_gold_kernel<<<CRF_B / 8, 256, 0, stream>>>(
        emissions, tags, mask, transitions, start_t, end_t, gold);
    crf_reduce_kernel<<<1, 1024, 0, stream>>>(fwd, gold, (float*)d_out);
}
